// MultiHeadAttention_86002425135549
// MI455X (gfx1250) — matrix operand order repaired and checked by reading, not compiled, not run on hardware
//
#include <hip/hip_runtime.h>
#include <hip/hip_bf16.h>
#include <stdint.h>

#define BATCH  4
#define SEQ    2048
#define NH     16
#define HD     64
#define HIDDEN 1024
#define NROWS  (BATCH*SEQ)   // 8192

typedef __attribute__((ext_vector_type(16))) __bf16 bf16x16;
typedef __attribute__((ext_vector_type(8)))  __bf16 bf16x8;
typedef __attribute__((ext_vector_type(8)))  float  f32x8;

static __device__ __forceinline__ f32x8 wmma_bf16(bf16x16 a, bf16x16 b, f32x8 c) {
  // D = A(16x32) * B(32x16) + C, f32 accumulate
  return __builtin_amdgcn_wmma_f32_16x16x32_bf16(false, a, false, b, (short)0, c,
                                                 false, false);
}

// Build a 16-element bf16 fragment from two 16-byte chunks (callers use contiguous k = 16h + s).
static __device__ __forceinline__ bf16x16 frag2(const __bf16* p0, const __bf16* p1) {
  bf16x8 lo = *(const bf16x8*)p0;
  bf16x8 hi = *(const bf16x8*)p1;
  bf16x16 r;
#pragma unroll
  for (int i = 0; i < 8; ++i) { r[i] = lo[i]; r[i + 8] = hi[i]; }
  return r;
}

// ---------------------------------------------------------------------------
// Kernel 1: QKV projection.  X[8192,1024] fp32 @ Wqkv[1024,3072] fp32 ->
// bf16 Q[b,h,s,d], K[b,h,s,d], Vt[b,h,d,s]  (V stored transposed for PV WMMA).
// Block tile 128x128, 8 waves (2 M x 4 N), wave tile 64x32 (4x2 WMMA tiles).
// ---------------------------------------------------------------------------
__global__ __launch_bounds__(256) void qkv_gemm_kernel(
    const float* __restrict__ X, const float* __restrict__ W,
    __bf16* __restrict__ Qb, __bf16* __restrict__ Kb, __bf16* __restrict__ Vt)
{
  constexpr int N = 3 * HIDDEN, Kd = HIDDEN;
  __shared__ __align__(16) __bf16 As[128][40];   // [m][k], pitch 80B (16B aligned)
  __shared__ __align__(16) __bf16 Bs[128][40];   // [n][k] (W transposed)
  const int bn = blockIdx.x % (N / 128), bm = blockIdx.x / (N / 128);
  const int m0 = bm * 128, n0 = bn * 128;
  const int tid = threadIdx.x, lane = tid & 31, wv = tid >> 5;
  const int wm = (wv & 1) * 64, wn = (wv >> 1) * 32;
  const int hf = lane >> 4, lr = lane & 15;
  f32x8 acc[4][2] = {};

  for (int k0 = 0; k0 < Kd; k0 += 32) {
    __syncthreads();
    { // stage X tile (fp32 -> bf16)
      int kc = (tid & 7) * 4, row = tid >> 3;
#pragma unroll
      for (int rp = 0; rp < 4; ++rp) {
        float4 v = *(const float4*)(X + (size_t)(m0 + row) * Kd + k0 + kc);
        __bf16* d = &As[row][kc];
        d[0] = (__bf16)v.x; d[1] = (__bf16)v.y; d[2] = (__bf16)v.z; d[3] = (__bf16)v.w;
        row += 32;
      }
      // stage W tile transposed: Bs[n][k]
      int nc = (tid & 31) * 4, kr = tid >> 5;
#pragma unroll
      for (int kp = 0; kp < 4; ++kp) {
        float4 v = *(const float4*)(W + (size_t)(k0 + kr) * N + n0 + nc);
        Bs[nc + 0][kr] = (__bf16)v.x; Bs[nc + 1][kr] = (__bf16)v.y;
        Bs[nc + 2][kr] = (__bf16)v.z; Bs[nc + 3][kr] = (__bf16)v.w;
        kr += 8;
      }
      if (k0 + 32 < Kd)  // hint next tile into caches
        __builtin_prefetch(X + (size_t)(m0 + (tid >> 3)) * Kd + k0 + 32, 0, 1);
    }
    __syncthreads();

    bf16x16 af[4], bfr[2];
#pragma unroll
    for (int i = 0; i < 4; ++i) {                 // A: lane row, 16 contiguous K
      const __bf16* p = &As[wm + i * 16 + lr][hf * 16];
      af[i] = frag2(p, p + 8);
    }
#pragma unroll
    for (int j = 0; j < 2; ++j) {                 // B: lane col, 16 contiguous K
      const __bf16* p = &Bs[wn + j * 16 + lr][hf * 16];
      bfr[j] = frag2(p, p + 8);
    }
#pragma unroll
    for (int i = 0; i < 4; ++i)
#pragma unroll
      for (int j = 0; j < 2; ++j)
        acc[i][j] = wmma_bf16(af[i], bfr[j], acc[i][j]);
  }

  // Epilogue: scatter into Q / K / Vt (C layout: lane = column, 8 rows).
#pragma unroll
  for (int i = 0; i < 4; ++i) {
#pragma unroll
    for (int j = 0; j < 2; ++j) {
      int n  = n0 + wn + j * 16 + lr;
      int t  = n / HIDDEN;            // 0=Q 1=K 2=V
      int hc = n % HIDDEN;
      int h  = hc >> 6, d = hc & 63;
      int mb = m0 + wm + i * 16 + hf * 8;
      int b  = mb / SEQ, s = mb % SEQ;
      size_t bh = (size_t)(b * NH + h);
      if (t == 2) {  // Vt[bh][d][s..s+7] is contiguous -> one 16B store
        __align__(16) __bf16 tmp[8];
#pragma unroll
        for (int r = 0; r < 8; ++r) tmp[r] = (__bf16)acc[i][j][r];
        *(uint4*)(Vt + (bh * HD + d) * SEQ + s) = *(const uint4*)tmp;
      } else {
        __bf16* dst = (t == 0 ? Qb : Kb) + (bh * SEQ + s) * HD + d;
#pragma unroll
        for (int r = 0; r < 8; ++r) dst[(size_t)r * HD] = (__bf16)acc[i][j][r];
      }
    }
  }
}

// ---------------------------------------------------------------------------
// Kernel 2: flash attention. One wave per (b,h, 16-query tile); loop over
// 32-key blocks; online softmax in C layout; P transposed C->A via LDS.
// ---------------------------------------------------------------------------
__global__ __launch_bounds__(256) void attn_kernel(
    const __bf16* __restrict__ Qb, const __bf16* __restrict__ Kb,
    const __bf16* __restrict__ Vt, const uint8_t* __restrict__ mask,
    __bf16* __restrict__ Ob)
{
  __shared__ __align__(16) __bf16 Psh[8][16][40];  // per-wave P tile [q][key]
  __shared__ __align__(16) __bf16 Osh[8][16][72];  // per-wave O stage [q][d]
  const int lane = threadIdx.x & 31, wv = threadIdx.x >> 5;
  const int job = blockIdx.x * 8 + wv;      // 64 bh * 128 qtiles = 8192 jobs
  const int qt = job & 127;
  const int bh = job >> 7;
  const int b = bh >> 4, h = bh & 15;
  const int hf = lane >> 4, lr = lane & 15;
  const __bf16* Qp = Qb + ((size_t)bh * SEQ + qt * 16) * HD;
  const __bf16* Kp = Kb + (size_t)bh * SEQ * HD;
  const __bf16* Vp = Vt + (size_t)bh * HD * SEQ;
  const uint8_t* mp = mask + b * SEQ;       // numpy bool: 1 byte/elem

  bf16x16 qf[2];                            // Q tile resident (d split 32+32)
#pragma unroll
  for (int ks = 0; ks < 2; ++ks) {
    const __bf16* p = Qp + (size_t)lr * HD + ks * 32 + hf * 16;
    qf[ks] = frag2(p, p + 8);
  }
  float mrow[8], lrow[8];
#pragma unroll
  for (int r = 0; r < 8; ++r) { mrow[r] = -1e30f; lrow[r] = 0.f; }
  f32x8 oacc[4] = {};
  const float sc = 0.125f * 1.44269504089f; // 1/sqrt(64) * log2(e)

  for (int kb = 0; kb < SEQ; kb += 32) {
    // --- scores: two 16-key tiles, K-dim = d (2 WMMA steps each) ---
    f32x8 s0 = {}, s1 = {};
#pragma unroll
    for (int ks = 0; ks < 2; ++ks) {
      const __bf16* pk0 = Kp + (size_t)(kb + lr) * HD + ks * 32 + hf * 16;
      s0 = wmma_bf16(qf[ks], frag2(pk0, pk0 + 8), s0);
      const __bf16* pk1 = Kp + (size_t)(kb + 16 + lr) * HD + ks * 32 + hf * 16;
      s1 = wmma_bf16(qf[ks], frag2(pk1, pk1 + 8), s1);
    }
    const float bias0 = mp[kb + lr]      ? 0.f : -1e30f;  // key = column = lane
    const float bias1 = mp[kb + 16 + lr] ? 0.f : -1e30f;
    float tm[8];
#pragma unroll
    for (int r = 0; r < 8; ++r) {
      float a = s0[r] * sc + bias0;
      float c = s1[r] * sc + bias1;
      s0[r] = a; s1[r] = c;
      tm[r] = fmaxf(a, c);
    }
    // row max across 16 lanes (xor<16 never crosses the half boundary)
#pragma unroll
    for (int off = 1; off < 16; off <<= 1)
#pragma unroll
      for (int r = 0; r < 8; ++r)
        tm[r] = fmaxf(tm[r], __shfl_xor(tm[r], off, 32));

    float tl[8];
#pragma unroll
    for (int r = 0; r < 8; ++r) {
      float mnew  = fmaxf(mrow[r], tm[r]);
      float alpha = exp2f(mrow[r] - mnew);
      mrow[r] = mnew;
      float p0 = exp2f(s0[r] - mnew);
      float p1 = exp2f(s1[r] - mnew);
      tl[r] = p0 + p1;
#pragma unroll
      for (int t = 0; t < 4; ++t) oacc[t][r] *= alpha;
      lrow[r] *= alpha;
      Psh[wv][hf * 8 + r][lr]      = (__bf16)p0;   // C layout -> row-major LDS
      Psh[wv][hf * 8 + r][16 + lr] = (__bf16)p1;
    }
#pragma unroll
    for (int off = 1; off < 16; off <<= 1)
#pragma unroll
      for (int r = 0; r < 8; ++r)
        tl[r] += __shfl_xor(tl[r], off, 32);
#pragma unroll
    for (int r = 0; r < 8; ++r) lrow[r] += tl[r];

    // same-wave LDS transpose: make stores visible before A-layout reads
    asm volatile("s_wait_dscnt 0x0" ::: "memory");
    const __bf16* pp = &Psh[wv][lr][hf * 16];      // A layout: row=lr, K=keys
    bf16x16 pf = frag2(pp, pp + 8);
#pragma unroll
    for (int t = 0; t < 4; ++t) {                  // O += P(16x32) * Vt(32x16)
      const __bf16* vp = Vp + (size_t)(t * 16 + lr) * SEQ + kb + hf * 16;
      oacc[t] = wmma_bf16(pf, frag2(vp, vp + 8), oacc[t]);
    }
  }

  // normalize, stage to LDS, then coalesced bf16 writes to Ob[b,s,h*64+d]
#pragma unroll
  for (int r = 0; r < 8; ++r) {
    float inv = 1.0f / lrow[r];
#pragma unroll
    for (int t = 0; t < 4; ++t)
      Osh[wv][hf * 8 + r][t * 16 + lr] = (__bf16)(oacc[t][r] * inv);
  }
  asm volatile("s_wait_dscnt 0x0" ::: "memory");
  {
    int row = lane >> 1, hh = lane & 1;
    const uint4* src = (const uint4*)&Osh[wv][row][hh * 32];
    __bf16* dst = Ob + ((size_t)b * SEQ + qt * 16 + row) * HIDDEN + h * HD + hh * 32;
#pragma unroll
    for (int i = 0; i < 4; ++i) ((uint4*)dst)[i] = src[i];
  }
}

// ---------------------------------------------------------------------------
// Kernel 3: output projection.  Ob[8192,1024] bf16 @ Wout[1024,1024] + bout
// -> out fp32.  Same GEMM structure as kernel 1.
// ---------------------------------------------------------------------------
__global__ __launch_bounds__(256) void out_gemm_kernel(
    const __bf16* __restrict__ A, const float* __restrict__ W,
    const float* __restrict__ bias, float* __restrict__ out)
{
  constexpr int N = HIDDEN, Kd = HIDDEN;
  __shared__ __align__(16) __bf16 As[128][40];
  __shared__ __align__(16) __bf16 Bs[128][40];
  const int bn = blockIdx.x % (N / 128), bm = blockIdx.x / (N / 128);
  const int m0 = bm * 128, n0 = bn * 128;
  const int tid = threadIdx.x, lane = tid & 31, wv = tid >> 5;
  const int wm = (wv & 1) * 64, wn = (wv >> 1) * 32;
  const int hf = lane >> 4, lr = lane & 15;
  f32x8 acc[4][2] = {};

  for (int k0 = 0; k0 < Kd; k0 += 32) {
    __syncthreads();
    { // stage A (already bf16): straight 16B copies
      int chunk = (tid & 3) * 8, row = tid >> 2;
#pragma unroll
      for (int rp = 0; rp < 2; ++rp) {
        uint4 v = *(const uint4*)(A + (size_t)(m0 + row) * Kd + k0 + chunk);
        *(uint4*)&As[row][chunk] = v;
        row += 64;
      }
      int nc = (tid & 31) * 4, kr = tid >> 5;
#pragma unroll
      for (int kp = 0; kp < 4; ++kp) {
        float4 v = *(const float4*)(W + (size_t)(k0 + kr) * N + n0 + nc);
        Bs[nc + 0][kr] = (__bf16)v.x; Bs[nc + 1][kr] = (__bf16)v.y;
        Bs[nc + 2][kr] = (__bf16)v.z; Bs[nc + 3][kr] = (__bf16)v.w;
        kr += 8;
      }
    }
    __syncthreads();

    bf16x16 af[4], bfr[2];
#pragma unroll
    for (int i = 0; i < 4; ++i) {
      const __bf16* p = &As[wm + i * 16 + lr][hf * 16];
      af[i] = frag2(p, p + 8);
    }
#pragma unroll
    for (int j = 0; j < 2; ++j) {
      const __bf16* p = &Bs[wn + j * 16 + lr][hf * 16];
      bfr[j] = frag2(p, p + 8);
    }
#pragma unroll
    for (int i = 0; i < 4; ++i)
#pragma unroll
      for (int j = 0; j < 2; ++j)
        acc[i][j] = wmma_bf16(af[i], bfr[j], acc[i][j]);
  }

#pragma unroll
  for (int i = 0; i < 4; ++i)
#pragma unroll
    for (int j = 0; j < 2; ++j) {
      int n  = n0 + wn + j * 16 + lr;
      float bv = bias[n];
      int mb = m0 + wm + i * 16 + hf * 8;
      float* dst = out + (size_t)mb * N + n;
#pragma unroll
      for (int r = 0; r < 8; ++r) dst[(size_t)r * N] = acc[i][j][r] + bv;
    }
}

// ---------------------------------------------------------------------------
extern "C" void kernel_launch(void* const* d_in, const int* in_sizes, int n_in,
                              void* d_out, int out_size, void* d_ws, size_t ws_size,
                              hipStream_t stream) {
  const float*   hs   = (const float*)d_in[0];
  const uint8_t* mask = (const uint8_t*)d_in[1];   // jnp.bool_: 1 byte/elem
  const float*   Wqkv = (const float*)d_in[2];
  const float*   Wout = (const float*)d_in[3];
  const float*   bout = (const float*)d_in[4];

  // Workspace: Q, K, Vt, O — each B*nh*S*64 bf16 = 16 MB (64 MB total).
  const size_t elems = (size_t)BATCH * NH * SEQ * HD;
  __bf16* Qb = (__bf16*)d_ws;
  __bf16* Kb = Qb + elems;
  __bf16* Vt = Kb + elems;
  __bf16* Ob = Vt + elems;

  qkv_gemm_kernel<<<dim3((NROWS / 128) * ((3 * HIDDEN) / 128)), dim3(256), 0, stream>>>(
      hs, Wqkv, Qb, Kb, Vt);
  attn_kernel<<<dim3((BATCH * NH * (SEQ / 16)) / 8), dim3(256), 0, stream>>>(
      Qb, Kb, Vt, mask, Ob);
  out_gemm_kernel<<<dim3((NROWS / 128) * (HIDDEN / 128)), dim3(256), 0, stream>>>(
      Ob, Wout, bout, (float*)d_out);
}
